// AdjConv_16011638079618
// MI455X (gfx1250) — compile-verified, matrix operand order checked by reading, not run on hardware
//
#include <hip/hip_runtime.h>

// ---------------------------------------------------------------------------
// CDNA5 (gfx1250) bf16 WMMA pipeline for the hypergraph-attention reference.
// wave32; v_wmma_f32_16x16x32_bf16 with fp32 accumulation; all tile traffic
// in 16-byte (b128) units; double-buffered LDS staging in the GEMM.
// ---------------------------------------------------------------------------

typedef __attribute__((ext_vector_type(16))) __bf16        v16bf;
typedef __attribute__((ext_vector_type(8)))  __bf16        v8bf;
typedef __attribute__((ext_vector_type(8)))  float         v8f;
typedef __attribute__((ext_vector_type(4)))  unsigned int  u32x4;

#define N_    4096
#define E_    1024
#define DIN_  256
#define H_    64
#define THETA 0.01f

// ---- bf16 helpers ---------------------------------------------------------
static __device__ inline unsigned short f32_to_bf16(float f) {
    unsigned u = __float_as_uint(f);
    u += 0x7FFFu + ((u >> 16) & 1u);   // round-to-nearest-even
    return (unsigned short)(u >> 16);
}
static __device__ inline __bf16 us_bf(unsigned short u) {
    union { unsigned short s; __bf16 b; } c; c.s = u; return c.b;
}
// two 16B-aligned groups of 8 bf16 -> one WMMA fragment register pair
static __device__ inline v16bf ld_frag16(const unsigned short* p0,
                                         const unsigned short* p1) {
    v8bf lo = *(const v8bf*)p0;
    v8bf hi = *(const v8bf*)p1;
    return __builtin_shufflevector(lo, hi, 0, 1, 2, 3, 4, 5, 6, 7,
                                           8, 9, 10, 11, 12, 13, 14, 15);
}
static __device__ inline v8f wmma_bf16(v16bf a, v16bf b, v8f c) {
    return __builtin_amdgcn_wmma_f32_16x16x32_bf16(
        false, a, false, b, (short)0, c, false, false);
}
static __device__ inline float wave_reduce_sum(float v) {
    for (int o = 16; o > 0; o >>= 1) v += __shfl_xor(v, o, 32);
    return v;
}

// ---------------------------------------------------------------------------
// Elementwise f32 -> bf16
// ---------------------------------------------------------------------------
__global__ void cvt_bf16(const float* __restrict__ src,
                         unsigned short* __restrict__ dst, int n) {
    for (int i = blockIdx.x * blockDim.x + threadIdx.x; i < n;
         i += gridDim.x * blockDim.x)
        dst[i] = f32_to_bf16(src[i]);
}

// ---------------------------------------------------------------------------
// Tiled transpose + convert: src (R x C) f32 -> dst (C x R) bf16
// ---------------------------------------------------------------------------
__global__ void transpose_cvt(const float* __restrict__ src,
                              unsigned short* __restrict__ dst, int R, int C) {
    __shared__ float tile[32][33];
    int tx = threadIdx.x, ty = threadIdx.y;
    int x = blockIdx.x * 32 + tx;
    for (int j = ty; j < 32; j += 8) {
        int y = blockIdx.y * 32 + j;
        if (y < R && x < C) tile[j][tx] = src[(size_t)y * C + x];
    }
    __syncthreads();
    int ox = blockIdx.y * 32 + tx;
    for (int j = ty; j < 32; j += 8) {
        int oy = blockIdx.x * 32 + j;
        if (oy < C && ox < R)
            dst[(size_t)oy * R + ox] = f32_to_bf16(tile[tx][j]);
    }
}

// ---------------------------------------------------------------------------
// Column / row sums
// ---------------------------------------------------------------------------
__global__ void colsum_k(const float* __restrict__ src,
                         float* __restrict__ out, int R, int C) {
    int c = blockIdx.x * blockDim.x + threadIdx.x;
    if (c >= C) return;
    float s = 0.f;
    for (int r = 0; r < R; ++r) s += src[(size_t)r * C + c];
    out[c] = s;
}

__global__ void rowsum_k(const float* __restrict__ src,
                         float* __restrict__ out, int C) {
    __shared__ float red[256];
    int r = blockIdx.x;
    float s = 0.f;
    for (int c = threadIdx.x; c < C; c += 256) s += src[(size_t)r * C + c];
    red[threadIdx.x] = s;
    __syncthreads();
    for (int off = 128; off > 0; off >>= 1) {
        if (threadIdx.x < off) red[threadIdx.x] += red[threadIdx.x + off];
        __syncthreads();
    }
    if (threadIdx.x == 0) out[r] = red[0];
}

// ---------------------------------------------------------------------------
// Row LayerNorm over H_=64 (optional pre-scale by 1/scale[row]); wave per row
// ---------------------------------------------------------------------------
__global__ void ln_rows(const float* __restrict__ X,
                        const float* __restrict__ scale,
                        const float* __restrict__ lw,
                        const float* __restrict__ lb,
                        float* __restrict__ Y, int rows) {
    int wave = threadIdx.x >> 5, lane = threadIdx.x & 31;
    int row = blockIdx.x * 8 + wave;
    if (row >= rows) return;
    float inv = scale ? 1.0f / scale[row] : 1.0f;
    float x0 = X[(size_t)row * H_ + lane] * inv;
    float x1 = X[(size_t)row * H_ + lane + 32] * inv;
    float m = wave_reduce_sum(x0 + x1) * (1.0f / H_);
    float d0 = x0 - m, d1 = x1 - m;
    float v = wave_reduce_sum(d0 * d0 + d1 * d1) * (1.0f / H_);
    float rs = rsqrtf(v + 1e-5f);
    Y[(size_t)row * H_ + lane]      = d0 * rs * lw[lane] + lb[lane];
    Y[(size_t)row * H_ + lane + 32] = d1 * rs * lw[lane + 32] + lb[lane + 32];
}

// ---------------------------------------------------------------------------
// sq[row] = sum_h X[row,h]^2 w[h];  sw[row,h] = bf16(X[row,h] w[h])
// ---------------------------------------------------------------------------
__global__ void rowstat_k(const float* __restrict__ X,
                          const float* __restrict__ w,
                          float* __restrict__ sq,
                          unsigned short* __restrict__ sw, int rows) {
    int wave = threadIdx.x >> 5, lane = threadIdx.x & 31;
    int row = blockIdx.x * 8 + wave;
    if (row >= rows) return;
    float x0 = X[(size_t)row * H_ + lane];
    float x1 = X[(size_t)row * H_ + lane + 32];
    float w0 = w[lane], w1 = w[lane + 32];
    float p = wave_reduce_sum(x0 * x0 * w0 + x1 * x1 * w1);
    if (lane == 0) sq[row] = p;
    if (sw) {
        sw[(size_t)row * H_ + lane]      = f32_to_bf16(x0 * w0);
        sw[(size_t)row * H_ + lane + 32] = f32_to_bf16(x1 * w1);
    }
}

// ---------------------------------------------------------------------------
// P'[n,e] = ta[e,n]*rsqrt(DV[n])*rsqrt(DE[e]); emits Ph (NxE) and PtH (ExN)
// ---------------------------------------------------------------------------
__global__ void make_P(const float* __restrict__ ta,
                       const float* __restrict__ DV,
                       const float* __restrict__ DE,
                       unsigned short* __restrict__ Ph,
                       unsigned short* __restrict__ PtH) {
    __shared__ float tile[32][33];
    int tx = threadIdx.x, ty = threadIdx.y;
    int n = blockIdx.x * 32 + tx;
    float dv = DV[n];
    float ivn = dv > 0.f ? rsqrtf(dv) : 0.f;
    for (int j = ty; j < 32; j += 8) {
        int e = blockIdx.y * 32 + j;
        float de = DE[e];
        float v = ta[(size_t)e * N_ + n] * ivn * (de > 0.f ? rsqrtf(de) : 0.f);
        PtH[(size_t)e * N_ + n] = f32_to_bf16(v);
        tile[j][tx] = v;
    }
    __syncthreads();
    int e2 = blockIdx.y * 32 + tx;
    for (int j = ty; j < 32; j += 8) {
        int n2 = blockIdx.x * 32 + j;
        Ph[(size_t)n2 * E_ + e2] = f32_to_bf16(tile[tx][j]);
    }
}

// ---------------------------------------------------------------------------
// Generic bf16 WMMA GEMM: C(MxN f32) = A(MxK) @ B(KxN), all dims divide tiles.
// Block 256 threads = 8 waves (4x2), block tile 128x64, wave tile 32x32
// (2x2 wmma), BK=32. Double-buffered LDS; B staged transposed so both A and B
// fragments are two ds_load_b128 each.
// mode 0: C = acc
// mode 1: C = exp(-(sa[m] + db[n] - 2*acc + cb[0]) / 800)
// mode 2: C = THETA*acc + (1-THETA)*G[m,n]
// ---------------------------------------------------------------------------
#define BM 128
#define BN 64
#define BK 32
#define ASTR 40   // padded K-stride of As rows (16B multiple)
#define BSTR 40   // padded K-stride of BsT rows

__global__ __launch_bounds__(256)
void gemm_bf16(const unsigned short* __restrict__ A,
               const unsigned short* __restrict__ B,
               float* __restrict__ C,
               int M, int N, int K, int mode,
               const float* __restrict__ sa, const float* __restrict__ db,
               const float* __restrict__ cbp, const float* __restrict__ G) {
    (void)M;
    __shared__ unsigned short As[2][BM * ASTR];   // 2 x 10.0 KB
    __shared__ unsigned short BsT[2][BN * BSTR];  // 2 x  5.0 KB

    int tid = threadIdx.x;
    int wave = tid >> 5, lane = tid & 31;
    int wm = wave & 3, wn = wave >> 2;
    int sub = lane >> 4, r = lane & 15;
    size_t bm0 = (size_t)blockIdx.x * BM, bn0 = (size_t)blockIdx.y * BN;

    // staging geometry: A tile 128x32 = 512 b128 chunks (2/thread),
    //                   B tile  32x64 = 256 b128 chunks (1/thread, scattered)
    int ar0 = tid >> 2, ac0 = tid & 3;            // chunk t=0: row, k-chunk
    int ar1 = (tid + 256) >> 2, ac1 = (tid + 256) & 3;
    int bk = tid >> 3, bn8 = tid & 7;             // B: k row, 8-wide n chunk

    u32x4 av0, av1, bv;
    union U4 { u32x4 v; unsigned short s[8]; };

    // prologue: tile 0
    av0 = *(const u32x4*)(A + (bm0 + ar0) * K + ac0 * 8);
    av1 = *(const u32x4*)(A + (bm0 + ar1) * K + ac1 * 8);
    bv  = *(const u32x4*)(B + (size_t)bk * N + bn0 + bn8 * 8);
    *(u32x4*)&As[0][ar0 * ASTR + ac0 * 8] = av0;
    *(u32x4*)&As[0][ar1 * ASTR + ac1 * 8] = av1;
    {
        U4 u; u.v = bv;
        for (int t = 0; t < 8; ++t)
            BsT[0][(bn8 * 8 + t) * BSTR + bk] = u.s[t];
    }
    __syncthreads();

    v8f acc[2][2] = {};
    int nk = K / BK;
    for (int kt = 0; kt < nk; ++kt) {
        int cur = kt & 1;
        if (kt + 1 < nk) {            // prefetch next tile into registers
            int k0 = (kt + 1) * BK;
            av0 = *(const u32x4*)(A + (bm0 + ar0) * K + k0 + ac0 * 8);
            av1 = *(const u32x4*)(A + (bm0 + ar1) * K + k0 + ac1 * 8);
            bv  = *(const u32x4*)(B + (size_t)(k0 + bk) * N + bn0 + bn8 * 8);
        }

        v16bf af[2], bfr[2];
        for (int i = 0; i < 2; ++i) {
            const unsigned short* p =
                &As[cur][(wm * 32 + i * 16 + r) * ASTR + sub * 8];
            af[i] = ld_frag16(p, p + 16);
        }
        for (int j2 = 0; j2 < 2; ++j2) {
            const unsigned short* p =
                &BsT[cur][(wn * 32 + j2 * 16 + r) * BSTR + sub * 8];
            bfr[j2] = ld_frag16(p, p + 16);
        }
        for (int i = 0; i < 2; ++i)
            for (int j2 = 0; j2 < 2; ++j2)
                acc[i][j2] = wmma_bf16(af[i], bfr[j2], acc[i][j2]);

        if (kt + 1 < nk) {
            int nxt = 1 - cur;
            *(u32x4*)&As[nxt][ar0 * ASTR + ac0 * 8] = av0;
            *(u32x4*)&As[nxt][ar1 * ASTR + ac1 * 8] = av1;
            U4 u; u.v = bv;
            for (int t = 0; t < 8; ++t)
                BsT[nxt][(bn8 * 8 + t) * BSTR + bk] = u.s[t];
        }
        __syncthreads();
    }

    // epilogue: C element j of lane -> row j + 8*sub, col r (per 16x16 tile)
    float cb = (mode == 1) ? cbp[0] : 0.f;
    for (int i = 0; i < 2; ++i)
        for (int j2 = 0; j2 < 2; ++j2)
            for (int j = 0; j < 8; ++j) {
                size_t gm = bm0 + wm * 32 + i * 16 + j + sub * 8;
                size_t gn = bn0 + wn * 32 + j2 * 16 + r;
                float v = acc[i][j2][j];
                if (mode == 1)
                    v = __expf(-(sa[gm] + db[gn] - 2.f * v + cb) * (1.f / 800.f));
                else if (mode == 2)
                    v = THETA * v + (1.f - THETA) * G[gm * N + gn];
                C[gm * N + gn] = v;
            }
}

// ---------------------------------------------------------------------------
// Fused attention: d_pre = softmax(f f^T / 8) @ f, f: (N_ x 64) bf16.
// One-pass streaming softmax (scores are O(20): exp in fp32 without max
// subtraction is safe). Block = 128 threads = 4 waves; wave owns 16 rows.
// Z (softmax denominator) is accumulated by an extra WMMA against an all-ones
// B matrix, which places Z in the same accumulator row layout as acc.
// ---------------------------------------------------------------------------
#define VSTR 72   // vt  row stride (32 keys x 64 feats), 16B multiple
#define TSTR 40   // vtT row stride (64 feats x 32 keys)
#define SSTR 40   // st  row stride (16 rows x 32 keys)

__global__ __launch_bounds__(128)
void attn_kernel(const unsigned short* __restrict__ fh,
                 float* __restrict__ dpre) {
    __shared__ unsigned short vt[32 * VSTR];      // keys row-major
    __shared__ unsigned short vtT[H_ * TSTR];     // keys feature-major
    __shared__ unsigned short st[4][16 * SSTR];   // exp(S) per wave

    int tid = threadIdx.x;
    int wave = tid >> 5, lane = tid & 31;
    int sub = lane >> 4, r = lane & 15;
    size_t row0 = (size_t)blockIdx.x * 64 + wave * 16;

    // Q fragments straight from global (two global_load_b128 each)
    const unsigned short* qp = fh + (row0 + r) * H_;
    v16bf qf[2];
    qf[0] = ld_frag16(qp + sub * 8,      qp + 16 + sub * 8);
    qf[1] = ld_frag16(qp + 32 + sub * 8, qp + 48 + sub * 8);

    v16bf ones;
    for (int j = 0; j < 16; ++j) ones[j] = us_bf(0x3F80);  // 1.0bf

    v8f acc[4] = {};
    v8f zac = {};
    union U4 { u32x4 v; unsigned short s[8]; };

    for (int ct = 0; ct < N_ / 32; ++ct) {
        // stage 32 key rows into both layouts (2 b128 chunks / thread)
        for (int t = 0; t < 2; ++t) {
            int i = tid + t * 128;
            int rr = i >> 3, c8 = i & 7;
            u32x4 v = *(const u32x4*)(fh + (size_t)(ct * 32 + rr) * H_ + c8 * 8);
            *(u32x4*)&vt[rr * VSTR + c8 * 8] = v;
            U4 u; u.v = v;
            for (int q = 0; q < 8; ++q)
                vtT[(c8 * 8 + q) * TSTR + rr] = u.s[q];
        }
        __syncthreads();

        // S = Q @ Ktile^T : B(kk,n) = vt[key n][kk]
        for (int nt = 0; nt < 2; ++nt) {
            int col = nt * 16 + r;
            v8f S = {};
            for (int ks = 0; ks < 2; ++ks) {
                const unsigned short* p = &vt[col * VSTR + ks * 32 + sub * 8];
                S = wmma_bf16(qf[ks], ld_frag16(p, p + 16), S);
            }
            for (int j = 0; j < 8; ++j)   // C elem j -> row j+8*sub
                st[wave][(j + sub * 8) * SSTR + col] =
                    f32_to_bf16(__expf(S[j] * 0.125f));
        }
        // wave-local LDS RAW: DS ops are in-order per wave (DScnt waits
        // inserted by compiler); no block barrier needed for st.
        const unsigned short* pp = &st[wave][r * SSTR + sub * 8];
        v16bf pa = ld_frag16(pp, pp + 16);

        zac = wmma_bf16(pa, ones, zac);               // Z += rowsum(P)
        for (int ht = 0; ht < 4; ++ht) {              // acc += P @ Vtile
            const unsigned short* p = &vtT[(ht * 16 + r) * TSTR + sub * 8];
            acc[ht] = wmma_bf16(pa, ld_frag16(p, p + 16), acc[ht]);
        }
        __syncthreads();   // before vt/vtT are overwritten
    }

    for (int ht = 0; ht < 4; ++ht)
        for (int j = 0; j < 8; ++j) {
            size_t rr = row0 + j + sub * 8;
            dpre[rr * H_ + ht * 16 + r] = acc[ht][j] / zac[j];
        }
}

// ---------------------------------------------------------------------------
// Host-side launch
// ---------------------------------------------------------------------------
extern "C" void kernel_launch(void* const* d_in, const int* in_sizes, int n_in,
                              void* d_out, int out_size, void* d_ws, size_t ws_size,
                              hipStream_t stream) {
    (void)in_sizes; (void)n_in; (void)out_size; (void)ws_size;
    const float* adj  = (const float*)d_in[0];
    const float* G    = (const float*)d_in[1];
    const float* feats= (const float*)d_in[2];
    const float* Wv   = (const float*)d_in[3];
    const float* Wl   = (const float*)d_in[4];
    const float* wo   = (const float*)d_in[5];
    const float* wob  = (const float*)d_in[6];
    const float* ln1w = (const float*)d_in[7];
    const float* ln1b = (const float*)d_in[8];
    const float* ln2w = (const float*)d_in[9];
    const float* ln2b = (const float*)d_in[10];
    float* out = (float*)d_out;

    char* p = (char*)d_ws;
    auto alloc = [&](size_t bytes) -> char* {
        char* q = p; p += (bytes + 255) & ~(size_t)255; return q;
    };
    unsigned short* featsh = (unsigned short*)alloc((size_t)N_ * DIN_ * 2);
    unsigned short* WvT    = (unsigned short*)alloc((size_t)DIN_ * H_ * 2);
    unsigned short* WlT    = (unsigned short*)alloc((size_t)DIN_ * H_ * 2);
    unsigned short* adjTh  = (unsigned short*)alloc((size_t)E_ * N_ * 2);
    float*          adjsum = (float*)alloc((size_t)E_ * 4);
    float*          f32b   = (float*)alloc((size_t)N_ * H_ * 4);
    float*          g32b   = (float*)alloc((size_t)N_ * H_ * 4);
    unsigned short* fh     = (unsigned short*)alloc((size_t)N_ * H_ * 2);
    unsigned short* gh     = (unsigned short*)alloc((size_t)N_ * H_ * 2);
    float*          spre   = (float*)alloc((size_t)E_ * H_ * 4);
    float*          sbuf   = (float*)alloc((size_t)E_ * H_ * 4);
    float*          dpre   = (float*)alloc((size_t)N_ * H_ * 4);
    float*          dbuf   = (float*)alloc((size_t)N_ * H_ * 4);
    unsigned short* dTh    = (unsigned short*)alloc((size_t)H_ * N_ * 2);
    unsigned short* swh    = (unsigned short*)alloc((size_t)E_ * H_ * 2);
    float*          sa     = (float*)alloc((size_t)E_ * 4);
    float*          db     = (float*)alloc((size_t)N_ * 4);
    float*          ta     = (float*)alloc((size_t)E_ * N_ * 4);
    float*          DV     = (float*)alloc((size_t)N_ * 4);
    float*          DE     = (float*)alloc((size_t)E_ * 4);
    unsigned short* Ph     = (unsigned short*)alloc((size_t)N_ * E_ * 2);
    unsigned short* PtH    = (unsigned short*)alloc((size_t)E_ * N_ * 2);

    // 1) conversions / transposes / degree
    cvt_bf16<<<(N_ * DIN_ + 255) / 256, 256, 0, stream>>>(feats, featsh, N_ * DIN_);
    transpose_cvt<<<dim3(DIN_ / 32, H_ / 32), dim3(32, 8), 0, stream>>>(Wv, WvT, H_, DIN_);
    transpose_cvt<<<dim3(DIN_ / 32, H_ / 32), dim3(32, 8), 0, stream>>>(Wl, WlT, H_, DIN_);
    transpose_cvt<<<dim3(E_ / 32, N_ / 32), dim3(32, 8), 0, stream>>>(adj, adjTh, N_, E_);
    colsum_k<<<(E_ + 255) / 256, 256, 0, stream>>>(adj, adjsum, N_, E_);

    // 2) f = feats@Wv.T, g = feats@Wl.T   (4096x64, K=256)
    gemm_bf16<<<dim3(N_ / BM, H_ / BN), 256, 0, stream>>>(
        featsh, WvT, f32b, N_, H_, DIN_, 0, nullptr, nullptr, nullptr, nullptr);
    gemm_bf16<<<dim3(N_ / BM, H_ / BN), 256, 0, stream>>>(
        featsh, WlT, g32b, N_, H_, DIN_, 0, nullptr, nullptr, nullptr, nullptr);
    cvt_bf16<<<(N_ * H_ + 255) / 256, 256, 0, stream>>>(f32b, fh, N_ * H_);
    cvt_bf16<<<(N_ * H_ + 255) / 256, 256, 0, stream>>>(g32b, gh, N_ * H_);

    // 3) s_pre = adj.T @ g  (1024x64, K=4096); s = LN(s_pre / adjsum[row])
    gemm_bf16<<<dim3(E_ / BM, H_ / BN), 256, 0, stream>>>(
        adjTh, gh, spre, E_, H_, N_, 0, nullptr, nullptr, nullptr, nullptr);
    ln_rows<<<E_ / 8, 256, 0, stream>>>(spre, adjsum, ln1w, ln1b, sbuf, E_);

    // 4) fused softmax attention, then LN
    attn_kernel<<<N_ / 64, 128, 0, stream>>>(fh, dpre);
    ln_rows<<<N_ / 8, 256, 0, stream>>>(dpre, nullptr, ln2w, ln2b, dbuf, N_);

    // 5) row stats + (s*w) bf16 + d transpose
    rowstat_k<<<E_ / 8, 256, 0, stream>>>(sbuf, wo, sa, swh, E_);
    rowstat_k<<<N_ / 8, 256, 0, stream>>>(dbuf, wo, db, nullptr, N_);
    transpose_cvt<<<dim3(H_ / 32, N_ / 32), dim3(32, 8), 0, stream>>>(dbuf, dTh, N_, H_);

    // 6) ta = exp(-(sa + db - 2*(s*w)@d.T + b)/800)   (1024x4096, K=64)
    gemm_bf16<<<dim3(E_ / BM, N_ / BN), 256, 0, stream>>>(
        swh, dTh, ta, E_, N_, H_, 1, sa, db, wob, nullptr);

    // 7) hypergraph normalization factors and P'
    rowsum_k<<<E_, 256, 0, stream>>>(ta, DE, N_);
    colsum_k<<<N_ / 256, 256, 0, stream>>>(ta, DV, E_, N_);
    make_P<<<dim3(N_ / 32, E_ / 32), dim3(32, 8), 0, stream>>>(ta, DV, DE, Ph, PtH);

    // 8) out = (1-THETA)*G + THETA * (P' @ P'.T)   (4096x4096, K=1024)
    gemm_bf16<<<dim3(N_ / BM, N_ / BN), 256, 0, stream>>>(
        Ph, PtH, out, N_, N_, E_, 2, nullptr, nullptr, nullptr, G);
}
